// FisheyeConv_32152125178107
// MI455X (gfx1250) — matrix operand order repaired and checked by reading, not compiled, not run on hardware
//
#include <hip/hip_runtime.h>
#include <hip/hip_bf16.h>
#include <stdint.h>

typedef _Float16 v16h __attribute__((ext_vector_type(16)));
typedef float    v8f  __attribute__((ext_vector_type(8)));

// ---- WMMA fragment builders from LDS rows (K-contiguous rows of 32 halfs) ----
// A-matrix 16x32 f16 layout: lane L holds M=L&15; halfs 0..7 = K=kb0..kb0+7,
// halfs 8..15 = K=16+kb0..16+kb0+7 where kb0 = (L>>4)*8.
__device__ inline v16h fragA(const _Float16* row, int hi) {
    union { v16h v; uint2 u[4]; } f;
    const uint2* p0 = (const uint2*)(row + hi * 8);
    const uint2* p1 = (const uint2*)(row + 16 + hi * 8);
    f.u[0] = p0[0]; f.u[1] = p0[1]; f.u[2] = p1[0]; f.u[3] = p1[1];
    return f.v;
}
// B-matrix 32x16 f16 layout: lane L holds N=L&15; halfs 0..7 = K=kb0..kb0+7,
// halfs 8..15 = K=16+kb0..16+kb0+7 where kb0 = (L>>4)*8.
__device__ inline v16h fragB(const _Float16* row, int hi) {
    union { v16h v; uint2 u[4]; } f;
    const uint2* p0 = (const uint2*)(row + hi * 8);
    const uint2* p1 = (const uint2*)(row + 16 + hi * 8);
    f.u[0] = p0[0]; f.u[1] = p0[1]; f.u[2] = p1[0]; f.u[3] = p1[1];
    return f.v;
}

__device__ inline float silu_f(float v) { return v / (1.0f + __expf(-v)); }

// LDS-relative byte offset of a __shared__ object (generic -> AS(3) -> int).
__device__ inline unsigned lds_offset(const void* p) {
    return (unsigned)(uintptr_t)(__attribute__((address_space(3))) const void*)p;
}

// Async global->LDS copy of 16 bytes per lane (CDNA5 GLOBAL_LOAD_ASYNC_TO_LDS_B128,
// GVS mode: 64-bit SGPR base + 32-bit per-lane byte offset). Tracked by ASYNCcnt.
__device__ inline void async_copy_b128(unsigned lds_off, unsigned gvoff, const void* sbase) {
    asm volatile("global_load_async_to_lds_b128 %0, %1, %2"
                 :: "v"(lds_off), "v"(gvoff), "s"(sbase)
                 : "memory");
}
__device__ inline void wait_async0() {
    asm volatile("s_wait_asynccnt 0x0" ::: "memory");
}

// ---- K0: Wl (c=32,d=128,o=256) f32 -> WlT[o][q=c*128+d] f16 ----
__global__ void k_wlT(const float* __restrict__ wl, _Float16* __restrict__ WlT) {
    int t = blockIdx.x * 256 + threadIdx.x;          // 1048576
    int o = t >> 12, q = t & 4095;
    WlT[(size_t)o * 4096 + q] = (_Float16)wl[(size_t)q * 256 + o];
}

// ---- K1: 2x2 avg pool -> vals_h[b][n][c] f16 ----
__global__ void k_pool(const float* __restrict__ x, _Float16* __restrict__ vals_h) {
    int t = blockIdx.x * 256 + threadIdx.x;          // 262144
    int c = t & 127;
    int n = (t >> 7) & 1023;
    int b = t >> 17;
    int v = n >> 5, u = n & 31;
    const float* px = x + (((size_t)(b * 128 + c) * 64 + 2 * v) * 64 + 2 * u);
    float s = 0.25f * (px[0] + px[1] + px[64] + px[65]);
    vals_h[t] = (_Float16)s;
}

// ---- K2: lift rotations R[m][9] f32, m = n*4 + l ----
__global__ void k_rot(float* __restrict__ Rbuf) {
    int m = blockIdx.x * 256 + threadIdx.x;          // 4096
    int n = m >> 2, l = m & 3;
    int u = n & 31, v = n >> 5;
    float du = (float)u - 15.5f, dv = (float)v - 15.5f;
    float r = sqrtf(du * du + dv * dv);
    float theta = r * (1.0f / 16.0f);
    float phi = atan2f(dv, du);
    float st = sinf(theta), ct = cosf(theta);
    float kx = -sinf(phi), ky = cosf(phi);
    float oc = 1.0f - ct;
    float RA[9];
    RA[0] = 1.0f - oc * ky * ky; RA[1] = oc * kx * ky;        RA[2] = st * ky;
    RA[3] = oc * kx * ky;        RA[4] = 1.0f - oc * kx * kx; RA[5] = -st * kx;
    RA[6] = -st * ky;            RA[7] = st * kx;             RA[8] = 1.0f - oc * (kx * kx + ky * ky);
    float alpha = 1.57079632679489662f * (float)l;   // 2*pi*l/4
    float ca = cosf(alpha), sa = sinf(alpha);
    float* R = Rbuf + (size_t)m * 9;
#pragma unroll
    for (int i = 0; i < 3; i++) {
        R[i * 3 + 0] =  RA[i * 3 + 0] * ca + RA[i * 3 + 1] * sa;
        R[i * 3 + 1] = -RA[i * 3 + 0] * sa + RA[i * 3 + 1] * ca;
        R[i * 3 + 2] =  RA[i * 3 + 2];
    }
}

// ---- K3: KNN-32 by largest gram (== smallest geodesic dist), tie -> lower idx ----
__global__ void k_topk(const float* __restrict__ Rbuf, int* __restrict__ idxbuf) {
    __shared__ float g[4096];
    __shared__ float sval[256];
    __shared__ int   sidx[256];
    int m = blockIdx.x, t = threadIdx.x;
    float rm[9];
#pragma unroll
    for (int i = 0; i < 9; i++) rm[i] = Rbuf[(size_t)m * 9 + i];
    for (int j = t; j < 4096; j += 256) {
        const float* rj = Rbuf + (size_t)j * 9;
        float s = 0.0f;
#pragma unroll
        for (int i = 0; i < 9; i++) s += rm[i] * rj[i];
        g[j] = s;
    }
    __syncthreads();
    for (int r = 0; r < 32; r++) {
        float bv = -1e30f; int bi = 0x7fffffff;
        for (int j = t; j < 4096; j += 256) {
            float vv = g[j];
            if (vv > bv || (vv == bv && j < bi)) { bv = vv; bi = j; }
        }
        sval[t] = bv; sidx[t] = bi;
        __syncthreads();
        for (int s = 128; s > 0; s >>= 1) {
            if (t < s) {
                float v2 = sval[t + s]; int i2 = sidx[t + s];
                if (v2 > sval[t] || (v2 == sval[t] && i2 < sidx[t])) { sval[t] = v2; sidx[t] = i2; }
            }
            __syncthreads();
        }
        if (t == 0) { idxbuf[m * 32 + r] = sidx[0]; g[sidx[0]] = -1e30f; }
        __syncthreads();
    }
}

// ---- K4: rel rotation + so3_log + 3->32->32 MLP; fold 1/KNN; store Wk f16 (m,k,c) ----
__global__ void k_mlp(const float* __restrict__ Rbuf, const int* __restrict__ idxbuf,
                      const float* __restrict__ w1, const float* __restrict__ b1,
                      const float* __restrict__ w2, const float* __restrict__ b2,
                      _Float16* __restrict__ Wk_h) {
    __shared__ float w1s[96], b1s[32], w2s[1024], b2s[32];
    int t = threadIdx.x;
    for (int i = t; i < 96; i += 256) w1s[i] = w1[i];
    if (t < 32) { b1s[t] = b1[t]; b2s[t] = b2[t]; }
    for (int i = t; i < 1024; i += 256) w2s[i] = w2[i];
    __syncthreads();
    int gid = blockIdx.x * 256 + t;                  // 131072 = 4096*32
    int m = gid >> 5;
    const float* Rm = Rbuf + (size_t)m * 9;
    const float* Rn = Rbuf + (size_t)idxbuf[gid] * 9;
    float rel[9];
#pragma unroll
    for (int a = 0; a < 3; a++)
#pragma unroll
        for (int bb = 0; bb < 3; bb++) {
            float s = 0.0f;
#pragma unroll
            for (int c = 0; c < 3; c++) s += Rm[c * 3 + a] * Rn[c * 3 + bb];
            rel[a * 3 + bb] = s;
        }
    float tr = rel[0] + rel[4] + rel[8];
    float cc = (tr - 1.0f) * 0.5f;
    cc = fminf(fmaxf(cc, -1.0f + 1e-6f), 1.0f - 1e-6f);
    float th = acosf(cc);
    float sn = sinf(th);
    float scale = (th < 1e-4f) ? 0.5f : th / (2.0f * sn);
    float e0 = scale * (rel[7] - rel[5]);
    float e1 = scale * (rel[2] - rel[6]);
    float e2 = scale * (rel[3] - rel[1]);
    float hm[32];
#pragma unroll
    for (int j = 0; j < 32; j++) {
        float h = e0 * w1s[j] + e1 * w1s[32 + j] + e2 * w1s[64 + j] + b1s[j];
        hm[j] = silu_f(h);
    }
    _Float16* wo = Wk_h + (size_t)gid * 32;
#pragma unroll 4
    for (int c = 0; c < 32; c++) {
        float acc = b2s[c];
#pragma unroll
        for (int j = 0; j < 32; j++) acc += hm[j] * w2s[j * 32 + c];
        wo[c] = (_Float16)(acc * 0.03125f);          // fold 1/KNN
    }
}

// ---- K5 (WMMA GEMM-A): per m: P(32x128) = Wk^T(32x32) x Vn(32x128), f16 out ----
__global__ void k_gemmA(const _Float16* __restrict__ Wk_h, const _Float16* __restrict__ vals_h,
                        const int* __restrict__ idxbuf, _Float16* __restrict__ partial,
                        int b, int mstart) {
    __shared__ __attribute__((aligned(16))) _Float16 As[32][32];   // [c][k]
    __shared__ __attribute__((aligned(16))) _Float16 Vt[128][32];  // [d][k]
    int t = threadIdx.x;                              // 128 threads
    int m = mstart + blockIdx.x;
    for (int i = t; i < 1024; i += 128) {             // transpose Wk (k,c) -> As[c][k]
        int k = i >> 5, c = i & 31;
        As[c][k] = Wk_h[((size_t)m * 32 + k) * 32 + c];
    }
    {
        int k = t >> 2;
        int d0 = (t & 3) * 32;
        int src = idxbuf[m * 32 + k] >> 2;            // repeat(vals, LIFT): V[idx] = vals[idx/4]
        const _Float16* vp = vals_h + ((size_t)(b * 1024 + src)) * 128 + d0;
#pragma unroll
        for (int j = 0; j < 32; j++) Vt[d0 + j][k] = vp[j];
    }
    __syncthreads();
    int w = t >> 5, L = t & 31, Llo = L & 15, Lhi = L >> 4;
    int dbase = w * 32;
    _Float16* pp = partial + (size_t)blockIdx.x * 32 * 128;
#pragma unroll
    for (int ct = 0; ct < 2; ct++) {
        v16h a = fragA(&As[ct * 16 + Llo][0], Lhi);
#pragma unroll
        for (int dt = 0; dt < 2; dt++) {
            v16h bf = fragB(&Vt[dbase + dt * 16 + Llo][0], Lhi);
            v8f cacc = {};
            cacc = __builtin_amdgcn_wmma_f32_16x16x32_f16(false, a, false, bf,
                                                          (short)0, cacc, false, false);
            int dcol = dbase + dt * 16 + Llo;
#pragma unroll
            for (int vg = 0; vg < 8; vg++) {
                int crow = ct * 16 + Lhi * 8 + vg;    // D layout: lanes>=16 hold M=8..15
                pp[(size_t)crow * 128 + dcol] = (_Float16)cacc[vg];
            }
        }
    }
}

// ---- K6 (WMMA GEMM-B + LIFT pool), async double-buffered staging ----
// 32 rows x 256 cols per workgroup: Pflat(32x4096) x WlT(4096x256); pool l=0..3, +bl.
// A+B panels stream via GLOBAL_LOAD_ASYNC_TO_LDS_B128 (ASYNCcnt), ping-pong LDS.
__global__ void k_gemmB(const _Float16* __restrict__ partial, const _Float16* __restrict__ WlT,
                        const float* __restrict__ bl, float* __restrict__ ybuf,
                        int b, int mstart) {
    __shared__ __attribute__((aligned(16))) _Float16 As[2][32][32];   // [buf][row][k]
    __shared__ __attribute__((aligned(16))) _Float16 Bs[2][256][32];  // [buf][o][k]
    int t = threadIdx.x;                              // 128 threads = 4 waves
    int w = t >> 5, L = t & 31, Llo = L & 15, Lhi = L >> 4;
    int rowbase = blockIdx.x * 32;                    // chunk-local row base
    unsigned AsBase = lds_offset(&As[0][0][0]);
    unsigned BsBase = lds_offset(&Bs[0][0][0]);
    int rA = t >> 2, segA = t & 3;                    // A panel: 32 rows x 4 x 16B

    auto issue_batch = [&](int kt) {
        int buf = kt & 1;
        int c = kt >> 2, dstart = (kt & 3) * 32;
        // A: 2KB = 128 lanes x 16B from partial[(rowbase+r)*32 + c][dstart + seg*8 ..]
        unsigned gA = (unsigned)(((((rowbase + rA) * 32 + c) * 128) + dstart + segA * 8) * 2);
        unsigned lA = AsBase + (unsigned)(buf * 2048 + rA * 64 + segA * 16);
        async_copy_b128(lA, gA, partial);
        // B: 16KB = 8 x (128 lanes x 16B) from WlT[o][kt*32 ..]
#pragma unroll
        for (int j = 0; j < 8; j++) {
            int chunk = j * 128 + t;
            int o = chunk >> 2, part = chunk & 3;
            unsigned gB = (unsigned)((o * 4096 + kt * 32) * 2 + part * 16);
            unsigned lB = BsBase + (unsigned)(buf * 16384 + o * 64 + part * 16);
            async_copy_b128(lB, gB, WlT);
        }
    };

    v8f acc[2][4] = {};
    issue_batch(0);
    for (int kt = 0; kt < 128; kt++) {
        wait_async0();                                // own wave's batch kt landed in LDS
        __syncthreads();                              // all waves' batch kt visible
        if (kt + 1 < 128) issue_batch(kt + 1);        // prefetch into other buffer
        int buf = kt & 1;
#pragma unroll
        for (int rt = 0; rt < 2; rt++) {
            v16h a = fragA(&As[buf][rt * 16 + Llo][0], Lhi);
#pragma unroll
            for (int nt = 0; nt < 4; nt++) {
                v16h bf = fragB(&Bs[buf][w * 64 + nt * 16 + Llo][0], Lhi);
                acc[rt][nt] = __builtin_amdgcn_wmma_f32_16x16x32_f16(
                    false, a, false, bf, (short)0, acc[rt][nt], false, false);
            }
        }
    }
    // Pool over LIFT=4 within each 16-row D tile (rows 4r..4r+3 share a pixel n),
    // add bl, store pre-BN y in (B,256,32,32) order: ybuf[(b*256+o)*1024 + n].
#pragma unroll
    for (int rt = 0; rt < 2; rt++) {
        int nbase = ((mstart + rowbase) >> 2) + rt * 4;
#pragma unroll
        for (int nt = 0; nt < 4; nt++) {
            int col = w * 64 + nt * 16 + Llo;
            float blv = bl[col];
            float s0 = (acc[rt][nt][0] + acc[rt][nt][1] + acc[rt][nt][2] + acc[rt][nt][3]) * 0.25f + blv;
            float s1 = (acc[rt][nt][4] + acc[rt][nt][5] + acc[rt][nt][6] + acc[rt][nt][7]) * 0.25f + blv;
            int n0 = nbase + Lhi * 2;                 // lanes 0-15 -> rows 0-7; 16-31 -> 8-15
            float* yp = ybuf + (size_t)(b * 256 + col) * 1024;
            yp[n0]     = s0;
            yp[n0 + 1] = s1;
        }
    }
}

// ---- K7: per-channel BN (population var) + affine + SiLU -> d_out ----
__global__ void k_bn(const float* __restrict__ ybuf, const float* __restrict__ gamma,
                     const float* __restrict__ beta, float* __restrict__ out) {
    __shared__ float ssum[256], ssq[256];
    __shared__ float mu_s, inv_s;
    int o = blockIdx.x, t = threadIdx.x;
    float s = 0.0f, q = 0.0f;
    for (int j = t; j < 2048; j += 256) {
        int bb = j >> 10, n = j & 1023;
        float v = ybuf[(size_t)(bb * 256 + o) * 1024 + n];
        s += v; q += v * v;
    }
    ssum[t] = s; ssq[t] = q;
    __syncthreads();
    for (int st = 128; st > 0; st >>= 1) {
        if (t < st) { ssum[t] += ssum[t + st]; ssq[t] += ssq[t + st]; }
        __syncthreads();
    }
    if (t == 0) {
        float mu = ssum[0] * (1.0f / 2048.0f);
        float var = ssq[0] * (1.0f / 2048.0f) - mu * mu;
        mu_s = mu; inv_s = rsqrtf(var + 1e-5f);
    }
    __syncthreads();
    float mu = mu_s, inv = inv_s, g = gamma[o], be = beta[o];
    for (int j = t; j < 2048; j += 256) {
        int bb = j >> 10, n = j & 1023;
        size_t idx = (size_t)(bb * 256 + o) * 1024 + n;
        float v = (ybuf[idx] - mu) * inv * g + be;
        out[idx] = silu_f(v);
    }
}

extern "C" void kernel_launch(void* const* d_in, const int* in_sizes, int n_in,
                              void* d_out, int out_size, void* d_ws, size_t ws_size,
                              hipStream_t stream) {
    const float* x     = (const float*)d_in[0];
    const float* w1    = (const float*)d_in[1];
    const float* b1    = (const float*)d_in[2];
    const float* w2    = (const float*)d_in[3];
    const float* b2    = (const float*)d_in[4];
    const float* wl    = (const float*)d_in[5];
    const float* bl    = (const float*)d_in[6];
    const float* gamma = (const float*)d_in[7];
    const float* beta  = (const float*)d_in[8];

    char* ws = (char*)d_ws;
    _Float16* vals_h  = (_Float16*)(ws + 0);          // 512 KB
    float*    Rbuf    = (float*)   (ws + 524288);     // 144 KB
    int*      idxbuf  = (int*)     (ws + 671744);     // 512 KB
    _Float16* Wk_h    = (_Float16*)(ws + 1196032);    // 8 MB
    _Float16* WlT     = (_Float16*)(ws + 9584640);    // 2 MB
    _Float16* partial = (_Float16*)(ws + 11681792);   // 16 MB (per-chunk, reused)
    float*    ybuf    = (float*)   (ws + 28459008);   // 2 MB
    float*    out     = (float*)d_out;                // (2,256,32,32)

    k_wlT <<<4096, 256, 0, stream>>>(wl, WlT);
    k_pool<<<1024, 256, 0, stream>>>(x, vals_h);
    k_rot <<<16,   256, 0, stream>>>(Rbuf);
    k_topk<<<4096, 256, 0, stream>>>(Rbuf, idxbuf);
    k_mlp <<<512,  256, 0, stream>>>(Rbuf, idxbuf, w1, b1, w2, b2, Wk_h);

    for (int b = 0; b < 2; b++)
        for (int ch = 0; ch < 2; ch++) {
            int mstart = ch * 2048;
            k_gemmA<<<2048, 128, 0, stream>>>(Wk_h, vals_h, idxbuf, partial, b, mstart);
            k_gemmB<<<64,   128, 0, stream>>>(partial, WlT, bl, ybuf, b, mstart);
        }

    k_bn<<<256, 256, 0, stream>>>(ybuf, gamma, beta, out);
}
